// HybridTokenMixer_2087354105895
// MI455X (gfx1250) — matrix operand order repaired and checked by reading, not compiled, not run on hardware
//
#include <hip/hip_runtime.h>
#include <hip/hip_bf16.h>
#include <math.h>

typedef _Float16 f16;
typedef __attribute__((ext_vector_type(16))) _Float16 v16h;
typedef __attribute__((ext_vector_type(8)))  _Float16 v8h;
typedef __attribute__((ext_vector_type(8)))  float    v8f;
typedef __attribute__((ext_vector_type(4)))  float    v4f;

#define BN_RS 0.99999500003749969f /* 1/sqrt(1+1e-5) */

// ---------------------------------------------------------------------------
// Dynamic-conv preparation: adaptive 3x3 pool + mean pool + dc_proj (2 tiny
// GEMMs) + softmax over G=2 -> per-(b,c) depthwise 3x3 weights and bias.
// One block per batch.
// ---------------------------------------------------------------------------
__global__ __launch_bounds__(128)
void dyn_prep_kernel(const float* __restrict__ x,
                     const float* __restrict__ p1w, const float* __restrict__ bn1g,
                     const float* __restrict__ bn1b,
                     const float* __restrict__ p2w, const float* __restrict__ p2b,
                     const float* __restrict__ dcw, const float* __restrict__ dcb,
                     float* __restrict__ dynw, float* __restrict__ dynb)
{
    __shared__ float pooled[128][9];
    __shared__ float meanp[128];
    __shared__ float hid[32][9];
    __shared__ float hid2[32];
    __shared__ float sc[256][9];
    __shared__ float s2v[256];
    const int b = blockIdx.x, t = threadIdx.x;
    const int hs[3] = {0, 18, 37}, he[3] = {19, 38, 56};

    { // adaptive pool (K=3, overlapping bins) + global mean, channel t
        const int c = t;
        const float* inp = x + (long)(b * 256 + c) * 3136;
        float total = 0.f;
        for (int s = 0; s < 3136; ++s) total += inp[s];
        meanp[c] = total * (1.0f / 3136.0f);
        for (int ih = 0; ih < 3; ++ih)
            for (int iw = 0; iw < 3; ++iw) {
                float acc = 0.f; int cnt = 0;
                for (int yy = hs[ih]; yy < he[ih]; ++yy)
                    for (int xx = hs[iw]; xx < he[iw]; ++xx) { acc += inp[yy * 56 + xx]; ++cnt; }
                pooled[c][ih * 3 + iw] = acc / (float)cnt;
            }
    }
    __syncthreads();
    // proj1: 32x128, relu(bn(.))
    for (int idx = t; idx < 32 * 9; idx += 128) {
        int r = idx / 9, s = idx % 9;
        float acc = 0.f;
        for (int c = 0; c < 128; ++c) acc += p1w[r * 128 + c] * pooled[c][s];
        acc = acc * (bn1g[r] * BN_RS) + bn1b[r];
        hid[r][s] = fmaxf(acc, 0.f);
    }
    if (t < 32) {
        float acc = 0.f;
        for (int c = 0; c < 128; ++c) acc += p1w[t * 128 + c] * meanp[c];
        acc = acc * (bn1g[t] * BN_RS) + bn1b[t];
        hid2[t] = fmaxf(acc, 0.f);
    }
    __syncthreads();
    // proj2: 256x32 + bias
    for (int idx = t; idx < 256 * 9; idx += 128) {
        int o = idx / 9, s = idx % 9;
        float acc = p2b[o];
        for (int r = 0; r < 32; ++r) acc += p2w[o * 32 + r] * hid[r][s];
        sc[o][s] = acc;
    }
    for (int o = t; o < 256; o += 128) {
        float acc = p2b[o];
        for (int r = 0; r < 32; ++r) acc += p2w[o * 32 + r] * hid2[r];
        s2v[o] = acc;
    }
    __syncthreads();
    // softmax over G=2, combine with dc_weight / dc_bias
    for (int idx = t; idx < 128 * 9; idx += 128) {
        int c = idx / 9, s = idx % 9;
        float a0 = sc[c][s], a1 = sc[128 + c][s];
        float m = fmaxf(a0, a1);
        float e0 = __expf(a0 - m), e1 = __expf(a1 - m);
        float w0 = e0 / (e0 + e1);
        dynw[((long)(b * 128) + c) * 9 + s] =
            w0 * dcw[c * 9 + s] + (1.f - w0) * dcw[(128 + c) * 9 + s];
    }
    if (t < 128) {
        float a0 = s2v[t], a1 = s2v[128 + t];
        float m = fmaxf(a0, a1);
        float e0 = __expf(a0 - m), e1 = __expf(a1 - m);
        float w0 = e0 / (e0 + e1);
        dynb[b * 128 + t] = w0 * dcb[t] + (1.f - w0) * dcb[128 + t];
    }
}

// ---------------------------------------------------------------------------
// Dynamic depthwise 3x3 (pad 1) with per-(b,c) weights -> xc channels 0..127
// ---------------------------------------------------------------------------
__global__ __launch_bounds__(256)
void dyn_dw_kernel(const float* __restrict__ x, const float* __restrict__ dynw,
                   const float* __restrict__ dynb, float* __restrict__ xc)
{
    int idx = blockIdx.x * 256 + threadIdx.x;
    if (idx >= 8 * 128 * 3136) return;
    int s = idx % 3136; int tmp = idx / 3136;
    int c = tmp % 128;  int b = tmp / 128;
    int h = s / 56, w = s % 56;
    const float* wv  = dynw + (long)(b * 128 + c) * 9;
    const float* inp = x + (long)(b * 256 + c) * 3136;
    float acc = dynb[b * 128 + c];
    #pragma unroll
    for (int kh = 0; kh < 3; ++kh) {
        int ih = h + kh - 1; if (ih < 0 || ih >= 56) continue;
        #pragma unroll
        for (int kw = 0; kw < 3; ++kw) {
            int iw = w + kw - 1; if (iw < 0 || iw >= 56) continue;
            acc += wv[kh * 3 + kw] * inp[ih * 56 + iw];
        }
    }
    xc[(long)(b * 256 + c) * 3136 + s] = acc;
}

// ---------------------------------------------------------------------------
// Spatial reduction: dw5x5 stride2 pad2 -> relu(bn) -> dw1x1 -> bn  (-> kv0)
// ---------------------------------------------------------------------------
__global__ __launch_bounds__(256)
void sr_kernel(const float* __restrict__ x, const float* __restrict__ w5,
               const float* __restrict__ g1, const float* __restrict__ b1,
               const float* __restrict__ s2w, const float* __restrict__ g2,
               const float* __restrict__ b2, float* __restrict__ kv0)
{
    int idx = blockIdx.x * 256 + threadIdx.x;
    if (idx >= 8 * 128 * 784) return;
    int s = idx % 784; int tmp = idx / 784;
    int c = tmp % 128; int b = tmp / 128;
    int oh = s / 28, ow = s % 28;
    const float* inp = x + (long)(b * 256 + 128 + c) * 3136;
    float acc = 0.f;
    #pragma unroll
    for (int kh = 0; kh < 5; ++kh) {
        int ih = oh * 2 - 2 + kh; if (ih < 0 || ih >= 56) continue;
        #pragma unroll
        for (int kw = 0; kw < 5; ++kw) {
            int iw = ow * 2 - 2 + kw; if (iw < 0 || iw >= 56) continue;
            acc += w5[c * 25 + kh * 5 + kw] * inp[ih * 56 + iw];
        }
    }
    float v = fmaxf(acc * (g1[c] * BN_RS) + b1[c], 0.f);
    v = v * s2w[c];
    v = v * (g2[c] * BN_RS) + b2[c];
    kv0[idx] = v;
}

// local dw3x3 pad1 + bias + residual -> kv
__global__ __launch_bounds__(256)
void local_kernel(const float* __restrict__ kv0, const float* __restrict__ w3,
                  const float* __restrict__ lb, float* __restrict__ kvb)
{
    int idx = blockIdx.x * 256 + threadIdx.x;
    if (idx >= 8 * 128 * 784) return;
    int s = idx % 784; int tmp = idx / 784;
    int c = tmp % 128; int b = tmp / 128;
    int oh = s / 28, ow = s % 28;
    const float* inp = kv0 + (long)(b * 128 + c) * 784;
    float acc = lb[c];
    #pragma unroll
    for (int kh = 0; kh < 3; ++kh) {
        int ih = oh + kh - 1; if (ih < 0 || ih >= 28) continue;
        #pragma unroll
        for (int kw = 0; kw < 3; ++kw) {
            int iw = ow + kw - 1; if (iw < 0 || iw >= 28) continue;
            acc += w3[c * 9 + kh * 3 + kw] * inp[ih * 28 + iw];
        }
    }
    kvb[idx] = acc + inp[oh * 28 + ow];
}

// ---------------------------------------------------------------------------
// Generic 1x1-conv GEMM via v_wmma_f32_16x16x32_f16.
// LDS tiles are stored in WMMA *fragment order* so each lane fetches its whole
// 32-byte fragment with one aligned v16h load (2x ds_load_b128).
//   A element (m,k):  lane = m + 16*((k>>3)&1), slot = ((k>>4)<<3) + (k&7)
//   B element (k,n):  lane = n + 16*((k>>3)&1), slot = ((k>>4)<<3) + (k&7)
// Staging: thread = (row r = t>>2, quarter q = t&3) handles 8 contiguous
// elements -> 2x v4f (global_load_b128) per tile; for A the fragment-order
// destination is also contiguous -> single v8h ds_store_b128.
// mode bit0 = ReLU (before BN), bit1 = BN scale/shift, bit2 = residual add.
// Block 128 = 4 waves; each wave owns a 16x16 subtile of the 32x32 tile.
// ---------------------------------------------------------------------------
__global__ __launch_bounds__(128)
void gemm1x1_wmma(const float* __restrict__ in, const float* __restrict__ Wm,
                  const float* __restrict__ bias,
                  const float* __restrict__ bng, const float* __restrict__ bnb,
                  const float* __restrict__ resid, float* __restrict__ out,
                  int Ci, int S, int Co,
                  long in_bstride, long in_coff,
                  long out_bstride, long res_bstride, int mode)
{
    __shared__ alignas(32) f16 AF[2][32][16];
    __shared__ alignas(32) f16 BF[2][32][16];
    const int t = threadIdx.x;
    const int lane = t & 31;
    const int wv = t >> 5;
    const int s0  = blockIdx.x * 32;
    const int co0 = blockIdx.y * 32;
    const int b   = blockIdx.z;
    const long inb = (long)b * in_bstride + in_coff;
    const int msub = wv >> 1;
    const int nsub = wv & 1;
    // staging assignment: row r (0..31), quarter q (0..3) -> 8 contiguous cols
    const int r   = t >> 2;
    const int qq  = t & 3;
    const int cb  = qq * 8;
    const int sbase = s0 + cb;
    const bool full = (sbase + 8 <= S);
    v8f c0 = {};

    for (int k0 = 0; k0 < Ci; k0 += 32) {
        // ---- A tile: 8 contiguous k of weight row (co0+r) ----
        {
            const float* ap = &Wm[(long)(co0 + r) * Ci + k0 + cb];
            v4f a0 = *(const v4f*)(ap);
            v4f a1 = *(const v4f*)(ap + 4);
            v8h pack;
            #pragma unroll
            for (int j = 0; j < 4; ++j) { pack[j] = (f16)a0[j]; pack[4 + j] = (f16)a1[j]; }
            // lane (r&15)+16*(q&1) constant, slots (q>>1)*8..+7 contiguous
            *(v8h*)(&AF[r >> 4][(r & 15) + 16 * (qq & 1)][(qq >> 1) * 8]) = pack;
        }
        // ---- B tile: 8 contiguous s of input row k = k0+r ----
        {
            const float* bp = &in[inb + (long)(k0 + r) * S + sbase];
            f16 hb[8];
            if (full) {
                v4f b0v = *(const v4f*)(bp);
                v4f b1v = *(const v4f*)(bp + 4);
                #pragma unroll
                for (int j = 0; j < 4; ++j) { hb[j] = (f16)b0v[j]; hb[4 + j] = (f16)b1v[j]; }
            } else {
                #pragma unroll
                for (int j = 0; j < 8; ++j)
                    hb[j] = (f16)((sbase + j < S) ? bp[j] : 0.0f);
            }
            #pragma unroll
            for (int j = 0; j < 8; ++j) {
                int c = cb + j;
                BF[c >> 4][(c & 15) + 16 * ((r >> 3) & 1)][((r >> 4) << 3) + (r & 7)] = hb[j];
            }
        }
        __syncthreads();
        v16h a  = *(const v16h*)(&AF[msub][lane][0]);
        v16h bf = *(const v16h*)(&BF[nsub][lane][0]);
        c0 = __builtin_amdgcn_wmma_f32_16x16x32_f16(false, a, false, bf,
                                                    (short)0, c0, false, false);
        __syncthreads();
    }
    #pragma unroll
    for (int rr = 0; rr < 8; ++rr) {
        int M = msub * 16 + rr + ((lane & 16) ? 8 : 0);
        int N = nsub * 16 + (lane & 15);
        int co = co0 + M;
        int s  = s0 + N;
        if (s < S) {
            float v = c0[rr] + bias[co];
            if (mode & 1) v = fmaxf(v, 0.0f);
            if (mode & 2) v = v * (bng[co] * BN_RS) + bnb[co];
            if (mode & 4) v += resid[(long)b * res_bstride + (long)co * S + s];
            out[(long)b * out_bstride + (long)co * S + s] = v;
        }
    }
}

// ---------------------------------------------------------------------------
// Fused SR-attention: one wave per (b, h, 16-row tile of N=3136).
// S = Q K^T * scale + rpe kept in LDS (16 x 800 f32, zero-padded), two-pass
// softmax over m=784, then O = P V with two 16x16 accumulators (d=32).
// K and V are channel-major, so each lane owns one channel and streams rows
// with coalesced v4f (global_load_b128) reads into a fragment-ordered LDS
// staging tile; fragments come back as single aligned v16h loads.
// ---------------------------------------------------------------------------
__global__ __launch_bounds__(32)
void attn_kernel(const float* __restrict__ q, const float* __restrict__ kvp,
                 const float* __restrict__ rpe, float* __restrict__ xc)
{
    __shared__ alignas(32) float Sm[16][800];   // scores -> probabilities (f32)
    __shared__ alignas(32) f16   FR[2][32][16]; // fragment-order K/V staging
    const int nt = blockIdx.x, h = blockIdx.y, b = blockIdx.z;
    const int lane = threadIdx.x;
    const int n0 = nt * 16;
    const int col = lane & 15;
    const int hiA = (lane & 16) ? 8 : 0;   // A K-base / C row offset
    const float scale = 0.17677669529663687f; // 32^-0.5

    // Q fragment (A: 16 rows n x K=32 d), reused for all 49 m-tiles
    v16h aq;
    #pragma unroll
    for (int e = 0; e < 16; ++e) {
        int kd = 2 * hiA + e;
        aq[e] = (f16)q[(long)((b * 128) + h * 32 + kd) * 3136 + n0 + col];
    }
    // this lane's K / V channels (d = lane, 0..31)
    const float* krow = kvp + (long)((b * 256) + h * 32 + lane) * 784;
    const float* vrow = kvp + (long)((b * 256) + 128 + h * 32 + lane) * 784;
    const int hi_d = lane >> 4, e_d = lane & 15;

    // ---- S = Q K^T * scale + rpe ----
    for (int mt = 0; mt < 49; ++mt) {
        const int m0 = mt * 16;
        // stage K tile (32 d x 16 m): coalesced v4f reads of this lane's row
        v4f kv0 = *(const v4f*)(krow + m0);
        v4f kv1 = *(const v4f*)(krow + m0 + 4);
        v4f kv2 = *(const v4f*)(krow + m0 + 8);
        v4f kv3 = *(const v4f*)(krow + m0 + 12);
        #pragma unroll
        for (int j = 0; j < 4; ++j) {
            FR[0][(j)      + 16 * hi_d][e_d] = (f16)kv0[j];
            FR[0][(j + 4)  + 16 * hi_d][e_d] = (f16)kv1[j];
            FR[0][(j + 8)  + 16 * hi_d][e_d] = (f16)kv2[j];
            FR[0][(j + 12) + 16 * hi_d][e_d] = (f16)kv3[j];
        }
        __syncthreads();
        v16h bk = *(const v16h*)(&FR[0][lane][0]);
        v8f c = {};
        c = __builtin_amdgcn_wmma_f32_16x16x32_f16(false, aq, false, bk,
                                                   (short)0, c, false, false);
        #pragma unroll
        for (int rr = 0; rr < 8; ++rr) {
            int Mr = rr + hiA;
            Sm[Mr][m0 + col] =
                c[rr] * scale + rpe[(long)((h * 3136) + n0 + Mr) * 784 + m0 + col];
        }
        __syncthreads();
    }
    __syncthreads();
    { // softmax over m: lanes (l, l+16) split one row, combine via shfl_xor 16
        const int row = lane & 15;
        const int j0 = (lane & 16) ? 392 : 0;
        float mx = -3.0e38f;
        for (int j = 0; j < 392; ++j) mx = fmaxf(mx, Sm[row][j0 + j]);
        mx = fmaxf(mx, __shfl_xor(mx, 16, 32));
        float sum = 0.f;
        for (int j = 0; j < 392; ++j) {
            float e = __expf(Sm[row][j0 + j] - mx);
            Sm[row][j0 + j] = e; sum += e;
        }
        sum += __shfl_xor(sum, 16, 32);
        float inv = 1.f / sum;
        for (int j = 0; j < 392; ++j) Sm[row][j0 + j] *= inv;
        if (lane < 16) for (int j = 784; j < 800; ++j) Sm[lane][j] = 0.f;
    }
    __syncthreads();
    // ---- O = P @ V : K = 800 (25 chunks of 32), N = 32 d-cols (two tiles) ----
    v8f o0 = {}, o1 = {};
    for (int kt = 0; kt < 25; ++kt) {
        const int k0 = kt * 32;
        // stage V chunk (32 m x 32 d): lane streams 32 m of its channel
        f16 tmp[32];
        #pragma unroll
        for (int jj = 0; jj < 32; jj += 4) {
            int m = k0 + jj;
            v4f vv;
            if (m + 3 < 784) {
                vv = *(const v4f*)(vrow + m);
            } else {
                vv[0] = (m + 0 < 784) ? vrow[m + 0] : 0.f;
                vv[1] = (m + 1 < 784) ? vrow[m + 1] : 0.f;
                vv[2] = (m + 2 < 784) ? vrow[m + 2] : 0.f;
                vv[3] = (m + 3 < 784) ? vrow[m + 3] : 0.f;
            }
            tmp[jj + 0] = (f16)vv[0]; tmp[jj + 1] = (f16)vv[1];
            tmp[jj + 2] = (f16)vv[2]; tmp[jj + 3] = (f16)vv[3];
        }
        { // two contiguous 16-half runs -> 2x(2x ds_store_b128)
            v16h lo, hi;
            #pragma unroll
            for (int j = 0; j < 16; ++j) { lo[j] = tmp[(j & 7) + ((j >> 3) << 4)]; hi[j] = tmp[8 + (j & 7) + ((j >> 3) << 4)]; }
            *(v16h*)(&FR[hi_d][e_d][0])      = lo;  // rows m 0..7, 16..23
            *(v16h*)(&FR[hi_d][e_d + 16][0]) = hi;  // rows m 8..15, 24..31
        }
        __syncthreads();
        // A fragment = P: two contiguous 8-float runs per lane (aligned v4f)
        const float* prow = &Sm[col][0];
        v4f p0 = *(const v4f*)(prow + k0 + hiA);
        v4f p1 = *(const v4f*)(prow + k0 + hiA + 4);
        v4f p2 = *(const v4f*)(prow + k0 + hiA + 16);
        v4f p3 = *(const v4f*)(prow + k0 + hiA + 20);
        v16h ap;
        #pragma unroll
        for (int j = 0; j < 4; ++j) {
            ap[j]      = (f16)p0[j];
            ap[4 + j]  = (f16)p1[j];
            ap[8 + j]  = (f16)p2[j];
            ap[12 + j] = (f16)p3[j];
        }
        v16h b0 = *(const v16h*)(&FR[0][lane][0]);
        v16h b1 = *(const v16h*)(&FR[1][lane][0]);
        o0 = __builtin_amdgcn_wmma_f32_16x16x32_f16(false, ap, false, b0,
                                                    (short)0, o0, false, false);
        o1 = __builtin_amdgcn_wmma_f32_16x16x32_f16(false, ap, false, b1,
                                                    (short)0, o1, false, false);
        __syncthreads();
    }
    #pragma unroll
    for (int rr = 0; rr < 8; ++rr) {
        int n = n0 + rr + hiA;
        xc[(long)((b * 256) + 128 + h * 32 + col) * 3136 + n]      = o0[rr];
        xc[(long)((b * 256) + 128 + 16 + h * 32 + col) * 3136 + n] = o1[rr];
    }
}

// pw1: depthwise 3x3 pad1 on xc (C=256) + bias -> relu -> bn -> z1
__global__ __launch_bounds__(256)
void pw1_dw_kernel(const float* __restrict__ xc, const float* __restrict__ w,
                   const float* __restrict__ bias, const float* __restrict__ g,
                   const float* __restrict__ bb, float* __restrict__ z1)
{
    int idx = blockIdx.x * 256 + threadIdx.x;
    if (idx >= 8 * 256 * 3136) return;
    int s = idx % 3136; int tmp = idx / 3136;
    int c = tmp % 256;  int b = tmp / 256;
    int y = s / 56, xp = s % 56;
    const float* inp = xc + (long)(b * 256 + c) * 3136;
    float acc = bias[c];
    #pragma unroll
    for (int kh = 0; kh < 3; ++kh) {
        int ih = y + kh - 1; if (ih < 0 || ih >= 56) continue;
        #pragma unroll
        for (int kw = 0; kw < 3; ++kw) {
            int iw = xp + kw - 1; if (iw < 0 || iw >= 56) continue;
            acc += w[c * 9 + kh * 3 + kw] * inp[ih * 56 + iw];
        }
    }
    float v = fmaxf(acc, 0.f);
    z1[idx] = v * (g[c] * BN_RS) + bb[c];
}

extern "C" void kernel_launch(void* const* d_in, const int* in_sizes, int n_in,
                              void* d_out, int out_size, void* d_ws, size_t ws_size,
                              hipStream_t stream) {
    (void)in_sizes; (void)n_in; (void)out_size; (void)ws_size;
    const float* x    = (const float*)d_in[0];
    const float* rpe  = (const float*)d_in[1];
    const float* dcw  = (const float*)d_in[2];
    const float* dcb  = (const float*)d_in[3];
    const float* p1w  = (const float*)d_in[4];
    const float* bn1g = (const float*)d_in[5];
    const float* bn1b = (const float*)d_in[6];
    const float* p2w  = (const float*)d_in[7];
    const float* p2b  = (const float*)d_in[8];
    const float* qw   = (const float*)d_in[9];
    const float* qb   = (const float*)d_in[10];
    const float* kvw  = (const float*)d_in[11];
    const float* kvbias = (const float*)d_in[12];
    const float* sr1w = (const float*)d_in[13];
    const float* sr1g = (const float*)d_in[14];
    const float* sr1b = (const float*)d_in[15];
    const float* sr2w = (const float*)d_in[16];
    const float* sr2g = (const float*)d_in[17];
    const float* sr2b = (const float*)d_in[18];
    const float* locw = (const float*)d_in[19];
    const float* locb = (const float*)d_in[20];
    const float* pw1  = (const float*)d_in[21];
    const float* pb1  = (const float*)d_in[22];
    const float* pbn1g = (const float*)d_in[23];
    const float* pbn1b = (const float*)d_in[24];
    const float* pw2  = (const float*)d_in[25];
    const float* pb2  = (const float*)d_in[26];
    const float* pbn2g = (const float*)d_in[27];
    const float* pbn2b = (const float*)d_in[28];
    const float* pw3  = (const float*)d_in[29];
    const float* pb3  = (const float*)d_in[30];
    const float* pbn3g = (const float*)d_in[31];
    const float* pbn3b = (const float*)d_in[32];
    float* out = (float*)d_out;

    // workspace layout (floats)
    float* ws   = (float*)d_ws;
    float* xc   = ws;                         // 8*256*3136
    float* qbuf = xc   + 8L * 256 * 3136;     // 8*128*3136
    float* kv0  = qbuf + 8L * 128 * 3136;     // 8*128*784
    float* kvB  = kv0  + 8L * 128 * 784;      // 8*128*784
    float* kvp  = kvB  + 8L * 128 * 784;      // 8*256*784
    float* z1   = kvp  + 8L * 256 * 784;      // 8*256*3136
    float* z2   = z1   + 8L * 256 * 3136;     // 8*32*3136
    float* dynw = z2   + 8L * 32 * 3136;      // 8*128*9
    float* dynb = dynw + 8L * 128 * 9;        // 8*128

    // 1. dynamic-conv weight/bias generation
    dyn_prep_kernel<<<8, 128, 0, stream>>>(x, p1w, bn1g, bn1b, p2w, p2b,
                                           dcw, dcb, dynw, dynb);
    // 2. y1 = dynamic depthwise conv -> xc[0:128]
    dyn_dw_kernel<<<(8 * 128 * 3136 + 255) / 256, 256, 0, stream>>>(x, dynw, dynb, xc);
    // 3. kv path (depthwise chain)
    sr_kernel<<<(8 * 128 * 784 + 255) / 256, 256, 0, stream>>>(
        x, sr1w, sr1g, sr1b, sr2w, sr2g, sr2b, kv0);
    local_kernel<<<(8 * 128 * 784 + 255) / 256, 256, 0, stream>>>(kv0, locw, locb, kvB);
    // 4. q = 1x1 conv on x2 (WMMA GEMM)
    gemm1x1_wmma<<<dim3(98, 4, 8), 128, 0, stream>>>(
        x, qw, qb, nullptr, nullptr, nullptr, qbuf,
        128, 3136, 128, 256L * 3136, 128L * 3136, 128L * 3136, 0, 0);
    // 5. kvp = 1x1 conv 128->256 on kv (WMMA GEMM)
    gemm1x1_wmma<<<dim3(25, 8, 8), 128, 0, stream>>>(
        kvB, kvw, kvbias, nullptr, nullptr, nullptr, kvp,
        128, 784, 256, 128L * 784, 0, 256L * 784, 0, 0);
    // 6. fused attention -> xc[128:256]
    attn_kernel<<<dim3(196, 4, 8), 32, 0, stream>>>(qbuf, kvp, rpe, xc);
    // 7. pw1 depthwise + relu + bn -> z1
    pw1_dw_kernel<<<(8 * 256 * 3136 + 255) / 256, 256, 0, stream>>>(
        xc, pw1, pb1, pbn1g, pbn1b, z1);
    // 8. pw2: 256->32, relu then bn
    gemm1x1_wmma<<<dim3(98, 1, 8), 128, 0, stream>>>(
        z1, pw2, pb2, pbn2g, pbn2b, nullptr, z2,
        256, 3136, 32, 256L * 3136, 0, 32L * 3136, 0, 3);
    // 9. pw3: 32->256, bn, + residual xc -> final output
    gemm1x1_wmma<<<dim3(98, 8, 8), 128, 0, stream>>>(
        z2, pw3, pb3, pbn3g, pbn3b, xc, out,
        32, 3136, 256, 32L * 3136, 0, 256L * 3136, 256L * 3136, 6);
}
